// PPFGCN_41437844472002
// MI455X (gfx1250) — matrix operand order repaired and checked by reading, not compiled, not run on hardware
//
#include <hip/hip_runtime.h>
#include <hip/hip_bf16.h>

typedef __attribute__((ext_vector_type(16))) __bf16 v16bf;
typedef __attribute__((ext_vector_type(8)))  __bf16 v8bf;
typedef __attribute__((ext_vector_type(8)))  float  v8f;

// ---------- order-preserving f32 <-> u32 encoding for atomic max ----------
__device__ __forceinline__ unsigned enc_f32(float x) {
    unsigned b = __float_as_uint(x);
    return (b & 0x80000000u) ? ~b : (b | 0x80000000u);
}
__device__ __forceinline__ float dec_f32(unsigned u) {
    unsigned b = (u & 0x80000000u) ? (u ^ 0x80000000u) : ~u;
    return __uint_as_float(b);
}
#define ENC_NEG_INF 0x007FFFFFu   // enc(-inf)

// ---------- geometry ----------
__device__ __forceinline__ float angle3(float ax, float ay, float az,
                                        float bx, float by, float bz) {
    float cx = ay * bz - az * by;
    float cy = az * bx - ax * bz;
    float cz = ax * by - ay * bx;
    float cn = sqrtf(cx * cx + cy * cy + cz * cz);
    float dt = ax * bx + ay * by + az * bz;
    return atan2f(cn, dt);
}

__device__ __forceinline__ v8f wmma_bf16(v16bf a, v16bf b, v8f c) {
    return __builtin_amdgcn_wmma_f32_16x16x32_bf16(
        false, a, false, b, (short)0, c, false, false);
}

// =====================================================================
// Weight staging: pack W[fout,fin] (f32, row-major) into LDS as bf16
// B-fragments (32x16 per fragment), fragment-major: each fragment is
// 32 lanes x 16 bf16 contiguous per lane (1 KB).  Zero-padded at K>=FIN.
// Fragment index within a layer region: frag = nt*NCHUNKS + chunk.
// =====================================================================
template <int FIN, int NCHUNKS>
__device__ __forceinline__ void stage_B(const float* __restrict__ W,
                                        __bf16* __restrict__ dst, int tid) {
    constexpr int TOTAL = 4 * NCHUNKS * 32 * 8;   // u32-pair granules
    for (int i = tid; i < TOTAL; i += 128) {
        int w     = i & 7;
        int lane  = (i >> 3) & 31;
        int fr    = i >> 8;                 // nt*NCHUNKS + chunk
        int chunk = fr % NCHUNKS;
        int nt    = fr / NCHUNKS;
        int n = nt * 16 + (lane & 15);
        int k = chunk * 32 + ((lane >> 4) & 1) * 8 + (w & 3) * 2 + (w >> 2) * 16;
        float f0 = (k     < FIN) ? W[n * FIN + k]     : 0.0f;
        float f1 = (k + 1 < FIN) ? W[n * FIN + k + 1] : 0.0f;
        __bf16* p = dst + fr * 512 + lane * 16 + 2 * w;
        p[0] = (__bf16)f0;
        p[1] = (__bf16)f1;
    }
}

// Unconditional 32-byte fragment load (2x ds_load_b128).
__device__ __forceinline__ v16bf lds_B(const __bf16* __restrict__ base,
                                       int frag, int lane) {
    return *(const v16bf*)(base + frag * 512 + lane * 16);
}

// A-fragment (16x32 bf16) from a row-major bf16 [16][64] activation slab.
// Lane's K-groups {b..b+7} and {b+16..b+23} are row-contiguous -> 2x b128.
__device__ __forceinline__ v16bf lds_A(const __bf16* __restrict__ h,
                                       int lane, int chunk) {
    const int M   = lane & 15;
    const int hi4 = (lane >> 4) & 1;
    const __bf16* p = h + M * 64 + chunk * 32 + hi4 * 8;
    v8bf lo = *(const v8bf*)p;
    v8bf hi = *(const v8bf*)(p + 16);
    v16bf v;
#pragma unroll
    for (int i = 0; i < 8; ++i) { v[i] = lo[i]; v[8 + i] = hi[i]; }
    return v;
}

// C-layout (M = r + hi*8, N = nt*16 + lane%16) -> bf16 slab, bias (+ReLU).
__device__ __forceinline__ void store_h(__bf16* __restrict__ h, int lane, int nt,
                                        const v8f& c, float bv, bool relu) {
#pragma unroll
    for (int r = 0; r < 8; ++r) {
        int M = r + ((lane >> 4) & 1) * 8;
        float v = c[r] + bv;
        if (relu) v = fmaxf(v, 0.0f);
        h[M * 64 + nt * 16 + (lane & 15)] = (__bf16)v;
    }
}

// A-fragment built directly from the encoded aggregation buffer (node kernel).
__device__ __forceinline__ v16bf agg_A_frag(const unsigned* __restrict__ agg,
                                            int n0, int lane, int chunk) {
    const int M   = lane & 15;
    const int hi4 = (lane >> 4) & 1;
    const unsigned* row = agg + (unsigned)(n0 + M) * 64u;
    v16bf a;
#pragma unroll
    for (int r = 0; r < 8; ++r) {
        int k = chunk * 32 + ((r < 4) ? 2 * r : 16 + 2 * (r - 4)) + hi4 * 8;
        a[2 * r]     = (__bf16)dec_f32(row[k]);
        a[2 * r + 1] = (__bf16)dec_f32(row[k + 1]);
    }
    return a;
}

// =====================================================================
// Kernel 0: init aggregation buffer to encoded(-inf)
// =====================================================================
__global__ void ppf_init_kernel(unsigned* __restrict__ agg, int n) {
    int i = blockIdx.x * blockDim.x + threadIdx.x;
    if (i < n) agg[i] = ENC_NEG_INF;
}

// =====================================================================
// Kernel 1: fused edge pipeline (grid-stride over 16-edge tiles)
// =====================================================================
__global__ __launch_bounds__(128)
void ppf_edge_kernel(const float* __restrict__ x,
                     const long long* __restrict__ ei,
                     const float* __restrict__ w1, const float* __restrict__ b1,
                     const float* __restrict__ w2, const float* __restrict__ b2,
                     const float* __restrict__ w3, const float* __restrict__ b3,
                     unsigned* __restrict__ agg,
                     long long n_edges, int e_total, int n_tiles) {
    // LDS: 20 weight frags (20KB) + biases + 4 per-wave bf16 slabs (8KB) + dst
    __shared__ __align__(32) __bf16 s_wB[20 * 512];
    __shared__ float  s_bias[3 * 64];
    __shared__ __align__(16) __bf16 s_h[4][16 * 64];
    __shared__ int    s_dst[4][16];

    const int tid  = threadIdx.x;
    const int wave = tid >> 5;
    const int lane = tid & 31;

    // ---- one-time weight/bias staging (all threads), then barrier ----
    stage_B<7,  1>(w1, s_wB + 0 * 512, tid);   // frags 0..3
    stage_B<64, 2>(w2, s_wB + 4 * 512, tid);   // frags 4..11
    stage_B<64, 2>(w3, s_wB + 12 * 512, tid);  // frags 12..19
    if (tid < 64) {
        s_bias[tid]       = b1[tid];
        s_bias[64 + tid]  = b2[tid];
        s_bias[128 + tid] = b3[tid];
    }
    __syncthreads();

    __bf16* h = s_h[wave];
    const int nidx = lane & 15;

    for (int tile = blockIdx.x * 4 + wave; tile < n_tiles; tile += gridDim.x * 4) {
        const int t0 = tile * 16;

        // ---- per-edge message (lanes 0..15) ----
        if (lane < 16) {
            long long e = (long long)t0 + lane;
            int s, d;
            if (e < n_edges) { s = (int)ei[e]; d = (int)ei[n_edges + e]; }
            else             { s = d = (int)(e - n_edges); }
            s_dst[wave][lane] = d;
            float sx = x[3 * s + 0], sy = x[3 * s + 1], sz = x[3 * s + 2];
            float tx = x[3 * d + 0], ty = x[3 * d + 1], tz = x[3 * d + 2];
            float dx = sx - tx, dy = sy - ty, dz = sz - tz;
            __bf16* row = h + lane * 64;
            row[0] = (__bf16)sx;
            row[1] = (__bf16)sy;
            row[2] = (__bf16)sz;
            row[3] = (__bf16)sqrtf(dx * dx + dy * dy + dz * dz);
            row[4] = (__bf16)angle3(tx, ty, tz, dx, dy, dz);
            row[5] = (__bf16)angle3(sx, sy, sz, dx, dy, dz);
            row[6] = (__bf16)angle3(tx, ty, tz, sx, sy, sz);
#pragma unroll
            for (int k = 7; k < 32; ++k) row[k] = (__bf16)0.0f;
        }

        // ---- layer 1: K padded to 32 (zeros already in LDS) ----
        {
            v16bf a = lds_A(h, lane, 0);
            v8f c[4];
#pragma unroll
            for (int nt = 0; nt < 4; ++nt) {
                v8f t = {};
                c[nt] = wmma_bf16(a, lds_B(s_wB, nt, lane), t);
            }
#pragma unroll
            for (int nt = 0; nt < 4; ++nt)
                store_h(h, lane, nt, c[nt], s_bias[nt * 16 + nidx], true);
        }

        // ---- layer 2 ----
        {
            v16bf a0 = lds_A(h, lane, 0);
            v16bf a1 = lds_A(h, lane, 1);
            v8f c[4];
#pragma unroll
            for (int nt = 0; nt < 4; ++nt) {
                v8f t = {};
                t = wmma_bf16(a0, lds_B(s_wB, 4 + nt * 2 + 0, lane), t);
                t = wmma_bf16(a1, lds_B(s_wB, 4 + nt * 2 + 1, lane), t);
                c[nt] = t;
            }
#pragma unroll
            for (int nt = 0; nt < 4; ++nt)
                store_h(h, lane, nt, c[nt], s_bias[64 + nt * 16 + nidx], true);
        }

        // ---- layer 3 + encoded atomic scatter-max ----
        {
            v16bf a0 = lds_A(h, lane, 0);
            v16bf a1 = lds_A(h, lane, 1);
#pragma unroll
            for (int nt = 0; nt < 4; ++nt) {
                v8f t = {};
                t = wmma_bf16(a0, lds_B(s_wB, 12 + nt * 2 + 0, lane), t);
                t = wmma_bf16(a1, lds_B(s_wB, 12 + nt * 2 + 1, lane), t);
                float bv = s_bias[128 + nt * 16 + nidx];
#pragma unroll
                for (int r = 0; r < 8; ++r) {
                    int M = r + ((lane >> 4) & 1) * 8;
                    if (t0 + M < e_total) {
                        int d = s_dst[wave][M];
                        atomicMax(&agg[(unsigned)d * 64u + nt * 16 + nidx],
                                  enc_f32(t[r] + bv));
                    }
                }
            }
        }
    }
}

// =====================================================================
// Kernel 2: global MLP over aggregated node features
// =====================================================================
__global__ __launch_bounds__(128)
void ppf_node_kernel(const unsigned* __restrict__ agg,
                     const float* __restrict__ w1, const float* __restrict__ b1,
                     const float* __restrict__ w2, const float* __restrict__ b2,
                     const float* __restrict__ w3, const float* __restrict__ b3,
                     float* __restrict__ out, int n_nodes, int n_tiles) {
    __shared__ __align__(32) __bf16 s_wB[24 * 512];   // 3 layers x 8 frags
    __shared__ float  s_bias[3 * 64];
    __shared__ __align__(16) __bf16 s_h[4][16 * 64];

    const int tid  = threadIdx.x;
    const int wave = tid >> 5;
    const int lane = tid & 31;

    stage_B<64, 2>(w1, s_wB + 0 * 512, tid);
    stage_B<64, 2>(w2, s_wB + 8 * 512, tid);
    stage_B<64, 2>(w3, s_wB + 16 * 512, tid);
    if (tid < 64) {
        s_bias[tid]       = b1[tid];
        s_bias[64 + tid]  = b2[tid];
        s_bias[128 + tid] = b3[tid];
    }
    __syncthreads();

    __bf16* h = s_h[wave];
    const int nidx = lane & 15;

    for (int tile = blockIdx.x * 4 + wave; tile < n_tiles; tile += gridDim.x * 4) {
        const int n0 = tile * 16;

        // layer 1 (A decoded straight from agg)
        {
            v16bf a0 = agg_A_frag(agg, n0, lane, 0);
            v16bf a1 = agg_A_frag(agg, n0, lane, 1);
            v8f c[4];
#pragma unroll
            for (int nt = 0; nt < 4; ++nt) {
                v8f t = {};
                t = wmma_bf16(a0, lds_B(s_wB, nt * 2 + 0, lane), t);
                t = wmma_bf16(a1, lds_B(s_wB, nt * 2 + 1, lane), t);
                c[nt] = t;
            }
#pragma unroll
            for (int nt = 0; nt < 4; ++nt)
                store_h(h, lane, nt, c[nt], s_bias[nt * 16 + nidx], true);
        }
        // layer 2
        {
            v16bf a0 = lds_A(h, lane, 0);
            v16bf a1 = lds_A(h, lane, 1);
            v8f c[4];
#pragma unroll
            for (int nt = 0; nt < 4; ++nt) {
                v8f t = {};
                t = wmma_bf16(a0, lds_B(s_wB, 8 + nt * 2 + 0, lane), t);
                t = wmma_bf16(a1, lds_B(s_wB, 8 + nt * 2 + 1, lane), t);
                c[nt] = t;
            }
#pragma unroll
            for (int nt = 0; nt < 4; ++nt)
                store_h(h, lane, nt, c[nt], s_bias[64 + nt * 16 + nidx], true);
        }
        // layer 3 -> d_out
        {
            v16bf a0 = lds_A(h, lane, 0);
            v16bf a1 = lds_A(h, lane, 1);
#pragma unroll
            for (int nt = 0; nt < 4; ++nt) {
                v8f t = {};
                t = wmma_bf16(a0, lds_B(s_wB, 16 + nt * 2 + 0, lane), t);
                t = wmma_bf16(a1, lds_B(s_wB, 16 + nt * 2 + 1, lane), t);
                float bv = s_bias[128 + nt * 16 + nidx];
#pragma unroll
                for (int r = 0; r < 8; ++r) {
                    int M = r + ((lane >> 4) & 1) * 8;
                    if (n0 + M < n_nodes)
                        out[(unsigned)(n0 + M) * 64u + nt * 16 + nidx] = t[r] + bv;
                }
            }
        }
    }
}

// =====================================================================
extern "C" void kernel_launch(void* const* d_in, const int* in_sizes, int n_in,
                              void* d_out, int out_size, void* d_ws, size_t ws_size,
                              hipStream_t stream) {
    const float*     x  = (const float*)d_in[0];
    const long long* ei = (const long long*)d_in[1];
    const float* lw1 = (const float*)d_in[2];  const float* lb1 = (const float*)d_in[3];
    const float* lw2 = (const float*)d_in[4];  const float* lb2 = (const float*)d_in[5];
    const float* lw3 = (const float*)d_in[6];  const float* lb3 = (const float*)d_in[7];
    const float* gw1 = (const float*)d_in[8];  const float* gb1 = (const float*)d_in[9];
    const float* gw2 = (const float*)d_in[10]; const float* gb2 = (const float*)d_in[11];
    const float* gw3 = (const float*)d_in[12]; const float* gb3 = (const float*)d_in[13];

    const int       n_nodes = in_sizes[0] / 3;        // 50000
    const long long n_edges = in_sizes[1] / 2;        // 1600000
    const int       e_total = (int)n_edges + n_nodes; // + self loops

    unsigned* agg = (unsigned*)d_ws;                  // [N,64] encoded f32

    {
        int n = n_nodes * 64;
        ppf_init_kernel<<<(n + 255) / 256, 256, 0, stream>>>(agg, n);
    }
    {
        int tiles  = (e_total + 15) / 16;             // 103125
        int blocks = (tiles + 3) / 4;
        if (blocks > 4096) blocks = 4096;             // grid-stride, amortize staging
        ppf_edge_kernel<<<blocks, 128, 0, stream>>>(
            x, ei, lw1, lb1, lw2, lb2, lw3, lb3, agg, n_edges, e_total, tiles);
    }
    {
        int tiles  = (n_nodes + 15) / 16;             // 3125
        int blocks = (tiles + 3) / 4;
        if (blocks > 512) blocks = 512;
        ppf_node_kernel<<<blocks, 128, 0, stream>>>(
            agg, gw1, gb1, gw2, gb2, gw3, gb3, (float*)d_out, n_nodes, tiles);
    }
}
